// Discriminator_22411139350694
// MI455X (gfx1250) — matrix operand order repaired and checked by reading, not compiled, not run on hardware
//
#include <hip/hip_runtime.h>
#include <hip/hip_bf16.h>
#include <math.h>

typedef _Float16 f16;
typedef __attribute__((ext_vector_type(16))) _Float16 v16h;
typedef __attribute__((ext_vector_type(8)))  float    v8f;

#define BATCH   2048
#define SEQ     1024
#define NTOK    1023            // S-1 tokens per batch row
#define HIDDEN  64
#define LOGCLAMP (-23.025850929940457f)   // ln(1e-10)

// ---------------------------------------------------------------------------
// Stage 1: fused MLP + Hermite logits.
// grid = (8, 2048), block = 256 (8 waves). Each wave handles 16 consecutive
// tokens (one WMMA M=16 tile) of batch row blockIdx.y.
// Layer2 (64x64) via v_wmma_f32_16x16x32_f16: 2 K-tiles x 4 N-blocks = 8 WMMA.
// All weights are pre-packed once per block into LDS in the exact per-lane
// consumption layout, so the compute phase is contiguous ds_load_b128 only.
// ---------------------------------------------------------------------------
__global__ __launch_bounds__(256)
void mlp_logits_kernel(const float* __restrict__ ys,
                       const float* __restrict__ gW1, const float* __restrict__ gb1,
                       const float* __restrict__ gW2, const float* __restrict__ gb2,
                       const float* __restrict__ gWo, const float* __restrict__ gbo,
                       float* __restrict__ z_out)
{
    // B-tiles of W2 in f16, per-lane B layout in this file's contiguous K order: [tile8][lane][e]
    // tile8 = kt*4+nb; lane l holds column n = nb*16+(l&15), element e is
    // K = kt*32 + (l>=16 ? 16 : 0) + e.
    __shared__ __align__(32) f16 sW2h[8 * 32 * 16];      // 8 KB
    // Layer-1 weights in per-(half,ktile,e) order: {W1[0][j],W1[1][j],W1[2][j],b1[j]}
    __shared__ float4 sW1p[2 * 2 * 16];                  // 1 KB
    // Per-hidden-unit n: {Wout[n][0], Wout[n][1], Wout[n][2], b2[n]}
    __shared__ float4 sWop[HIDDEN];                      // 1 KB
    __shared__ float  sBo[4];

    const int tid = threadIdx.x;

    // ---- cooperative weight pre-pack (once per block) ---------------------
    {   // W2 -> f16 B-tile layout: one (tile, lane) pair per thread
        const int tile = tid >> 5;          // 0..7  (kt = tile>>2, nb = tile&3)
        const int l    = tid & 31;
        const int kt   = tile >> 2;
        const int nb   = tile & 3;
        const int n    = nb * 16 + (l & 15);
        const int kb   = kt * 32 + ((l >> 4) ? 16 : 0);
        f16* dst = &sW2h[(tile * 32 + l) * 16];
        #pragma unroll
        for (int e = 0; e < 16; ++e)
            dst[e] = (f16)gW2[(kb + e) * HIDDEN + n];
    }
    if (tid < 64) {   // W1/b1 pack: idx = half*32 + ktile*16 + e
        const int h = tid >> 5, t = (tid >> 4) & 1, e = tid & 15;
        const int j = e + h * 16 + t * 32;
        sW1p[tid] = make_float4(gW1[j], gW1[64 + j], gW1[128 + j], gb1[j]);
    }
    if (tid >= 64 && tid < 128) {   // Wout + b2 pack, indexed by n
        const int n = tid - 64;
        sWop[n] = make_float4(gWo[n * 3 + 0], gWo[n * 3 + 1], gWo[n * 3 + 2], gb2[n]);
    }
    if (tid < 3) sBo[tid] = gbo[tid];
    __syncthreads();

    const int wave  = tid >> 5;
    const int lane  = tid & 31;
    const int half  = lane >> 4;      // which K-half of the wave this lane feeds
    const int lm    = lane & 15;      // row (A) / column (B,C) index within tile
    const int b     = blockIdx.y;
    const int sbase = blockIdx.x * 128 + wave * 16;

    // --- token features (lanes l and l+16 load the same token lm) ----------
    const int s  = sbase + lm;
    const int sc = s > (NTOK - 1) ? (NTOK - 1) : s;     // clamp edge (masked later)
    const float* row = ys + (size_t)b * (SEQ * 2);
    const float x0 = row[2 * sc + 1];                   // values[b, s]
    const float tt = row[2 * sc + 2];                   // timestamps[b, s+1]
    const float xt = row[2 * sc + 3];                   // values[b, s+1]

    // --- layer 1 (3->64) in VALU, packed straight into WMMA A-tile layout --
    v16h a0, a1;
    const float4* w1h = &sW1p[half * 32];
    #pragma unroll
    for (int e = 0; e < 16; ++e) {
        const float4 wa = w1h[e];            // K-tile 0
        const float4 wb = w1h[16 + e];       // K-tile 1
        float h  = fmaf(xt, wa.x, fmaf(tt, wa.y, fmaf(x0, wa.z, wa.w)));
        float h2 = fmaf(xt, wb.x, fmaf(tt, wb.y, fmaf(x0, wb.z, wb.w)));
        a0[e] = (f16)(h  > 0.f ? h  : 0.f);
        a1[e] = (f16)(h2 > 0.f ? h2 : 0.f);
    }

    // --- per-lane output weights / layer-2 bias ---------------------------
    float wo0[4], wo1[4], wo2[4], bias2[4];
    #pragma unroll
    for (int nb = 0; nb < 4; ++nb) {
        const float4 w = sWop[nb * 16 + lm];
        wo0[nb] = w.x; wo1[nb] = w.y; wo2[nb] = w.z; bias2[nb] = w.w;
    }

    // --- layer 2 (64x64) via WMMA: 4 N-blocks x 2 K-tiles ------------------
    v8f acc[4];
    #pragma unroll
    for (int nb = 0; nb < 4; ++nb) {
        const v16h bt0 = *(const v16h*)&sW2h[((0 * 4 + nb) * 32 + lane) * 16];
        const v16h bt1 = *(const v16h*)&sW2h[((1 * 4 + nb) * 32 + lane) * 16];
        v8f c = {};
        c = __builtin_amdgcn_wmma_f32_16x16x32_f16(false, a0, false, bt0,
                                                   (short)0, c, false, false);
        c = __builtin_amdgcn_wmma_f32_16x16x32_f16(false, a1, false, bt1,
                                                   (short)0, c, false, false);
        #pragma unroll
        for (int r = 0; r < 8; ++r) {
            const float v = c[r] + bias2[nb];
            c[r] = v > 0.f ? v : 0.f;                  // relu(h2)
        }
        acc[nb] = c;
    }

    // --- layer 3 (64->3) folded with Hermite basis -------------------------
    // z[m] = sum_n h2[m,n]*(Wo[n,0] + Wo[n,1]*2x_m + Wo[n,2]*(4x_m^2-2)) + bout.herm
    #pragma unroll
    for (int r = 0; r < 8; ++r) {
        const int m = r + half * 8;            // C layout: VGPR r -> row m
        const float xm = __shfl(xt, m, 32);    // x_t of row m (owned by lane m)
        const float H1 = 2.f * xm;
        const float H2 = 4.f * xm * xm - 2.f;
        float p = 0.f;
        #pragma unroll
        for (int nb = 0; nb < 4; ++nb)
            p += acc[nb][r] * (wo0[nb] + wo1[nb] * H1 + wo2[nb] * H2);
        // reduce across the 16 lanes of this half-wave (masks stay in-half)
        p += __shfl_xor(p, 1, 32);
        p += __shfl_xor(p, 2, 32);
        p += __shfl_xor(p, 4, 32);
        p += __shfl_xor(p, 8, 32);
        if (lm == 0) {                         // lanes 0 and 16 write m and m+8
            const int sm = sbase + m;
            if (sm < NTOK)
                z_out[(size_t)sm * BATCH + b] =
                    p + sBo[0] + sBo[1] * H1 + sBo[2] * H2;
        }
    }
}

// ---------------------------------------------------------------------------
// Stage 2: per-column (over b=0..2047) logsumexp + clamped log-prob sum.
// grid = 1023 blocks, 256 threads; 8 values per thread kept in registers.
// partial[s] = sum_b max(z[s,b] - lse_s, ln(1e-10))
// ---------------------------------------------------------------------------
__global__ __launch_bounds__(256)
void column_softmax_kernel(const float* __restrict__ z, float* __restrict__ partial)
{
    const int s   = blockIdx.x;
    const int tid = threadIdx.x;
    const int wv  = tid >> 5;
    const int ln  = tid & 31;
    const float* col = z + (size_t)s * BATCH;

    float v[8];
    float mx = -INFINITY;
    #pragma unroll
    for (int i = 0; i < 8; ++i) {
        v[i] = col[tid + 256 * i];
        mx = fmaxf(mx, v[i]);
    }

    __shared__ float red[8];
    #pragma unroll
    for (int m = 16; m >= 1; m >>= 1) mx = fmaxf(mx, __shfl_xor(mx, m, 32));
    if (ln == 0) red[wv] = mx;
    __syncthreads();
    float bmax = red[0];
    #pragma unroll
    for (int i = 1; i < 8; ++i) bmax = fmaxf(bmax, red[i]);
    __syncthreads();

    float se = 0.f;
    #pragma unroll
    for (int i = 0; i < 8; ++i) se += expf(v[i] - bmax);
    #pragma unroll
    for (int m = 16; m >= 1; m >>= 1) se += __shfl_xor(se, m, 32);
    if (ln == 0) red[wv] = se;
    __syncthreads();
    float bse = 0.f;
    #pragma unroll
    for (int i = 0; i < 8; ++i) bse += red[i];
    const float lse = bmax + logf(bse);
    __syncthreads();

    float a = 0.f;
    #pragma unroll
    for (int i = 0; i < 8; ++i) {
        const float lp = v[i] - lse;           // log softmax, exact
        a += (lp > LOGCLAMP ? lp : LOGCLAMP);  // clip(p, 1e-10)
    }
    #pragma unroll
    for (int m = 16; m >= 1; m >>= 1) a += __shfl_xor(a, m, 32);
    if (ln == 0) red[wv] = a;
    __syncthreads();
    if (tid == 0) {
        float t = 0.f;
        #pragma unroll
        for (int i = 0; i < 8; ++i) t += red[i];
        partial[s] = t;
    }
}

// ---------------------------------------------------------------------------
// Stage 3: reduce 1023 column partials, divide by batch -> scalar output.
// ---------------------------------------------------------------------------
__global__ __launch_bounds__(256)
void final_reduce_kernel(const float* __restrict__ partial, float* __restrict__ out)
{
    const int tid = threadIdx.x;
    const int wv  = tid >> 5;
    const int ln  = tid & 31;
    float a = 0.f;
    for (int i = tid; i < NTOK; i += 256) a += partial[i];
    #pragma unroll
    for (int m = 16; m >= 1; m >>= 1) a += __shfl_xor(a, m, 32);
    __shared__ float red[8];
    if (ln == 0) red[wv] = a;
    __syncthreads();
    if (tid == 0) {
        float t = 0.f;
        #pragma unroll
        for (int i = 0; i < 8; ++i) t += red[i];
        out[0] = t * (1.0f / (float)BATCH);
    }
}

// ---------------------------------------------------------------------------
extern "C" void kernel_launch(void* const* d_in, const int* in_sizes, int n_in,
                              void* d_out, int out_size, void* d_ws, size_t ws_size,
                              hipStream_t stream)
{
    const float* ys   = (const float*)d_in[0];  // (2048, 1024, 2)
    const float* W1   = (const float*)d_in[1];  // (3, 64)
    const float* b1   = (const float*)d_in[2];  // (64,)
    const float* W2   = (const float*)d_in[3];  // (64, 64)
    const float* b2   = (const float*)d_in[4];  // (64,)
    const float* Wout = (const float*)d_in[5];  // (64, 3)
    const float* bout = (const float*)d_in[6];  // (3,)

    float* z       = (float*)d_ws;                 // [1023][2048] logits
    float* partial = z + (size_t)NTOK * BATCH;     // [1023] column sums

    dim3 grid1(8, BATCH);   // 8 s-tiles of 128 tokens x 2048 batch rows
    mlp_logits_kernel<<<grid1, 256, 0, stream>>>(ys, W1, b1, W2, b2, Wout, bout, z);
    column_softmax_kernel<<<NTOK, 256, 0, stream>>>(z, partial);
    final_reduce_kernel<<<1, 256, 0, stream>>>(partial, (float*)d_out);
}
